// GINConvNet_8220567404982
// MI455X (gfx1250) — matrix operand order repaired and checked by reading, not compiled, not run on hardware
//
#include <hip/hip_runtime.h>
#include <hip/hip_bf16.h>
#include <stdint.h>

#define N_NODES 256000
#define N_EDGES 2048000
#define NB      512
#define FXD     78
#define DIM     32
#define OUTD    128
#define PLEN    1000
#define SLEN    100
#define EMBD    128
#define NF      32
#define CONVL   121
#define BN_EPS  1e-5f

typedef __attribute__((ext_vector_type(16))) __bf16 bf16x16;
typedef __attribute__((ext_vector_type(8)))  __bf16 bf16x8;
typedef __attribute__((ext_vector_type(8)))  float  f32x8;

// A-fragment (16x32 bf16): lane holds M=lane&15, K-halves split by lane>>4.
// a[0..15] = A[m][k0 + half*16 + 0..15] (32B contiguous)
__device__ __forceinline__ bf16x16 ld_afrag(const __bf16* __restrict__ rowp, int k0, int half) {
    const __bf16* p = rowp + k0 + half * 16;
    bf16x8 lo = *reinterpret_cast<const bf16x8*>(p);
    bf16x8 hi = *reinterpret_cast<const bf16x8*>(p + 8);
    return __builtin_shufflevector(lo, hi, 0, 1, 2, 3, 4, 5, 6, 7, 8, 9, 10, 11, 12, 13, 14, 15);
}

// B-fragment (32x16 bf16) from Bt[N x K]: lane holds N=lane&15.
// b[0..15] = Bt[n][k0 + half*16 + 0..15] (32B contiguous)
__device__ __forceinline__ bf16x16 ld_bfrag(const __bf16* __restrict__ rowp, int k0, int half) {
    const __bf16* p = rowp + k0 + half * 16;
    bf16x8 lo = *reinterpret_cast<const bf16x8*>(p);
    bf16x8 hi = *reinterpret_cast<const bf16x8*>(p + 8);
    return __builtin_shufflevector(lo, hi, 0, 1, 2, 3, 4, 5, 6, 7, 8, 9, 10, 11, 12, 13, 14, 15);
}

__device__ __forceinline__ f32x8 wmma_bf16(bf16x16 a, bf16x16 b, f32x8 c) {
    return __builtin_amdgcn_wmma_f32_16x16x32_bf16(false, a, false, b, (short)0, c, false, false);
}

// ---------------------------------------------------------------- f32 -> bf16
__global__ void cvt_bf16_kernel(const float* __restrict__ in, __bf16* __restrict__ out, int n) {
    int i = blockIdx.x * blockDim.x + threadIdx.x;
    if (i < n) out[i] = (__bf16)in[i];
}

// ------------------------------------------------- edge scatter-add (segment_sum)
template <int D>
__global__ void edge_agg_kernel(const float* __restrict__ h, const int* __restrict__ src,
                                const int* __restrict__ dst, float* __restrict__ agg, int Ee) {
    int e = blockIdx.x * blockDim.x + threadIdx.x;
    if (e >= Ee) return;
    int s = src[e], d = dst[e];
    const float* hp = h + (size_t)s * D;
    float* ap = agg + (size_t)d * D;
#pragma unroll 8
    for (int f = 0; f < D; ++f) atomicAdd(ap + f, hp[f]);
}

// ---------------------------------------- GIN MLP + BN-stat partial reduction
template <int DIN>
__global__ void gin_mlp_kernel(const float* __restrict__ hin, const float* __restrict__ agg,
                               const float* __restrict__ w1, const float* __restrict__ b1,
                               const float* __restrict__ w2, const float* __restrict__ b2,
                               float* __restrict__ hout, float* __restrict__ stats, int Nn) {
    __shared__ float sW1[DIM * DIN];
    __shared__ float sW2[DIM * DIM];
    __shared__ float sB1[DIM], sB2[DIM], sSum[DIM], sSq[DIM];
    for (int i = threadIdx.x; i < DIM * DIN; i += blockDim.x) sW1[i] = w1[i];
    for (int i = threadIdx.x; i < DIM * DIM; i += blockDim.x) sW2[i] = w2[i];
    if (threadIdx.x < DIM) {
        sB1[threadIdx.x] = b1[threadIdx.x];
        sB2[threadIdx.x] = b2[threadIdx.x];
        sSum[threadIdx.x] = 0.f;
        sSq[threadIdx.x]  = 0.f;
    }
    __syncthreads();
    int n = blockIdx.x * blockDim.x + threadIdx.x;
    float t[DIM], o[DIM];
    if (n < Nn) {
        const float* hp = hin + (size_t)n * DIN;
        const float* ap = agg + (size_t)n * DIN;
        for (int j = 0; j < DIM; ++j) t[j] = sB1[j];
        for (int d = 0; d < DIN; ++d) {
            float v = hp[d] + ap[d];
            for (int j = 0; j < DIM; ++j) t[j] += sW1[j * DIN + d] * v;
        }
        for (int j = 0; j < DIM; ++j) t[j] = fmaxf(t[j], 0.f);
        for (int j = 0; j < DIM; ++j) o[j] = sB2[j];
        for (int k = 0; k < DIM; ++k) {
            float tv = t[k];
            for (int j = 0; j < DIM; ++j) o[j] += sW2[j * DIM + k] * tv;
        }
        for (int j = 0; j < DIM; ++j) {
            o[j] = fmaxf(o[j], 0.f);
            hout[(size_t)n * DIM + j] = o[j];
        }
    } else {
        for (int j = 0; j < DIM; ++j) o[j] = 0.f;
    }
    for (int j = 0; j < DIM; ++j) {
        atomicAdd(&sSum[j], o[j]);
        atomicAdd(&sSq[j],  o[j] * o[j]);
    }
    __syncthreads();
    if (threadIdx.x < DIM) {
        atomicAdd(&stats[threadIdx.x],       sSum[threadIdx.x]);
        atomicAdd(&stats[DIM + threadIdx.x], sSq[threadIdx.x]);
    }
}

// -------------------------------------------------------------- BN apply
__global__ void bn_apply_kernel(float* __restrict__ h, const float* __restrict__ stats,
                                const float* __restrict__ gamma, const float* __restrict__ beta,
                                int Nn) {
    int i = blockIdx.x * blockDim.x + threadIdx.x;
    if (i >= Nn * DIM) return;
    int f = i & (DIM - 1);
    float inv = 1.0f / (float)Nn;
    float mu  = stats[f] * inv;
    float var = stats[DIM + f] * inv - mu * mu;
    h[i] = (h[i] - mu) * rsqrtf(var + BN_EPS) * gamma[f] + beta[f];
}

// -------------------------------------------------------------- global add pool
__global__ void pool_add_kernel(const float* __restrict__ h, const int* __restrict__ batch,
                                float* __restrict__ g, int Nn) {
    int n = blockIdx.x * blockDim.x + threadIdx.x;
    if (n >= Nn) return;
    int b = batch[n];
#pragma unroll
    for (int f = 0; f < DIM; ++f) atomicAdd(&g[(size_t)b * DIM + f], h[(size_t)n * DIM + f]);
}

// ---------------------------------------------------------------------------
// Generic bf16 WMMA GEMM:  C[M x N] = A[M x K] * Bt[N x K]^T + bias (+relu)
// REQUIRES: M%32==0, N%64==0, K%32==0 (true for all shapes in this net).
// Block = 256 threads = 8 waves; wave tile 16x16; block tile 32(M) x 64(N).
// Dual accumulators (K unroll 2) for WMMA ILP.
// ---------------------------------------------------------------------------
__global__ void gemm_bf16_wmma(const __bf16* __restrict__ A, const __bf16* __restrict__ Bt,
                               const float* __restrict__ bias, float* __restrict__ C, int ldc,
                               int K, int relu) {
    int lane = threadIdx.x & 31;
    int wave = threadIdx.x >> 5;
    int half = lane >> 4;
    int m0 = blockIdx.y * 32 + (wave >> 2) * 16;
    int n0 = blockIdx.x * 64 + (wave & 3) * 16;
    const __bf16* Arow = A + (size_t)(m0 + (lane & 15)) * K;
    const __bf16* Brow = Bt + (size_t)(n0 + (lane & 15)) * K;
    f32x8 acc0 = {}, acc1 = {};
    int k0 = 0;
    for (; k0 + 64 <= K; k0 += 64) {
        acc0 = wmma_bf16(ld_afrag(Arow, k0, half), ld_bfrag(Brow, k0, half), acc0);
        acc1 = wmma_bf16(ld_afrag(Arow, k0 + 32, half), ld_bfrag(Brow, k0 + 32, half), acc1);
    }
    if (k0 < K)
        acc0 = wmma_bf16(ld_afrag(Arow, k0, half), ld_bfrag(Brow, k0, half), acc0);
    f32x8 acc = acc0 + acc1;
    int nrow = n0 + (lane & 15);
    float bv = bias[nrow];
#pragma unroll
    for (int v = 0; v < 8; ++v) {
        float val = acc[v] + bv;
        if (relu) val = fmaxf(val, 0.0f);
        C[(size_t)(m0 + v + half * 8) * ldc + nrow] = val;
    }
}

// ---------------------------------------------------------------------------
// Conv1d-as-implicit-GEMM with WMMA. One block per graph.
//   out[f,l] = sum_c sum_k8 E[token[c], l+k8] * w[f, c*8+k8]
// M=32 filters (2 tiles), N=128 spatial (8 tiles; store l<121), K=C*8.
// Embedding table + tokens staged in LDS; B-tiles gathered from LDS.
// Each 16-element B-fragment half covers exactly 2 channel groups (k8=0..7 x2).
// ---------------------------------------------------------------------------
__global__ void conv_pool_wmma(const int* __restrict__ tokens, const float* __restrict__ embf,
                               const __bf16* __restrict__ wA, const float* __restrict__ bias,
                               float* __restrict__ convout, float* __restrict__ pooled,
                               int C, int V) {
    __shared__ __bf16 sE[100 * EMBD];  // max table: 100 x 128
    __shared__ int    sT[PLEN];        // max tokens: 1000
    int b = blockIdx.x;
    for (int i = threadIdx.x; i < V * EMBD; i += blockDim.x) sE[i] = (__bf16)embf[i];
    for (int i = threadIdx.x; i < C; i += blockDim.x) sT[i] = tokens[(size_t)b * C + i];
    __syncthreads();

    int lane = threadIdx.x & 31;
    int wave = threadIdx.x >> 5;
    int half = lane >> 4;
    int mt  = wave >> 2;        // 0..1 -> filter tile
    int ntb = (wave & 3) * 2;   // two adjacent N tiles per wave
    int K = C * 8;
    const __bf16* wArow = wA + (size_t)(mt * 16 + (lane & 15)) * K;
    int ncol0 = ntb * 16 + (lane & 15);
    int ncol1 = ncol0 + 16;
    f32x8 acc0 = {}, acc1 = {};
    for (int k0 = 0; k0 < K; k0 += 32) {
        bf16x16 a = ld_afrag(wArow, k0, half);
        int c0 = (k0 + half * 16) >> 3;           // first of 2 channel groups
        const __bf16* r0 = sE + sT[c0] * EMBD;
        const __bf16* r1 = sE + sT[c0 + 1] * EMBD;
        bf16x16 bA = {}, bB = {};
#pragma unroll
        for (int j = 0; j < 8; ++j) {             // j == k8
            int p0 = ncol0 + j; if (p0 > 127) p0 = 127;   // clamped cols feed only
            int p1 = ncol1 + j; if (p1 > 127) p1 = 127;   // discarded outputs (l>=121)
            bA[j]     = r0[p0];
            bA[8 + j] = r1[p0];
            bB[j]     = r0[p1];
            bB[8 + j] = r1[p1];
        }
        acc0 = wmma_bf16(a, bA, acc0);
        acc1 = wmma_bf16(a, bB, acc1);
    }
#pragma unroll
    for (int v = 0; v < 8; ++v) {
        int f  = mt * 16 + v + half * 8;
        int l0 = ncol0;
        int l1 = ncol1;
        if (l0 < CONVL) {
            float val = fmaxf(acc0[v] + bias[f], 0.0f);
            convout[((size_t)b * NF + f) * CONVL + l0] = val;
            atomicAdd(&pooled[(size_t)b * NF + f], val);
        }
        if (l1 < CONVL) {
            float val = fmaxf(acc1[v] + bias[f], 0.0f);
            convout[((size_t)b * NF + f) * CONVL + l1] = val;
            atomicAdd(&pooled[(size_t)b * NF + f], val);
        }
    }
}

// ---------------------------------------------- SE block + flatten + cvt bf16
__global__ void se_flatten_kernel(const float* __restrict__ convout,
                                  const float* __restrict__ pooled,
                                  const float* __restrict__ w1, const float* __restrict__ b1,
                                  const float* __restrict__ w2, const float* __restrict__ b2,
                                  __bf16* __restrict__ outbf) {
    int b = blockIdx.x;
    __shared__ float s_h[2];
    __shared__ float s_scale[NF];
    if (threadIdx.x < 2) {
        float acc = b1[threadIdx.x];
        for (int f = 0; f < NF; ++f)
            acc += (pooled[(size_t)b * NF + f] * (1.0f / (float)CONVL)) * w1[threadIdx.x * NF + f];
        s_h[threadIdx.x] = fmaxf(acc, 0.0f);
    }
    __syncthreads();
    if (threadIdx.x < NF) {
        float acc = b2[threadIdx.x] + s_h[0] * w2[threadIdx.x * 2 + 0] +
                    s_h[1] * w2[threadIdx.x * 2 + 1];
        s_scale[threadIdx.x] = 1.0f / (1.0f + expf(-acc));
    }
    __syncthreads();
    for (int i = threadIdx.x; i < NF * CONVL; i += blockDim.x) {
        int f = i / CONVL;
        outbf[(size_t)b * (NF * CONVL) + i] =
            (__bf16)(convout[(size_t)b * (NF * CONVL) + i] * s_scale[f]);
    }
}

// -------------------------------------------------------------- final 256->1
__global__ void final_out_kernel(const float* __restrict__ x, const float* __restrict__ w,
                                 const float* __restrict__ bias, float* __restrict__ out) {
    int b = blockIdx.x * blockDim.x + threadIdx.x;
    if (b >= NB) return;
    float acc = bias[0];
    for (int k = 0; k < 256; ++k) acc += x[(size_t)b * 256 + k] * w[k];
    out[b] = acc;
}

// ===========================================================================
extern "C" void kernel_launch(void* const* d_in, const int* in_sizes, int n_in,
                              void* d_out, int out_size, void* d_ws, size_t ws_size,
                              hipStream_t stream) {
    (void)in_sizes; (void)n_in; (void)out_size; (void)ws_size;
    const float* x      = (const float*)d_in[0];
    const int*   ei     = (const int*)d_in[1];
    const int*   batch  = (const int*)d_in[2];
    const int*   target = (const int*)d_in[3];
    const int*   smiles = (const int*)d_in[4];
    const int* src = ei;
    const int* dst = ei + N_EDGES;

    const float *gin_w1[5], *gin_b1[5], *gin_w2[5], *gin_b2[5], *bn_g[5], *bn_b[5];
    for (int i = 0; i < 5; ++i) {
        int base = 5 + i * 6;
        gin_w1[i] = (const float*)d_in[base + 0];
        gin_b1[i] = (const float*)d_in[base + 1];
        gin_w2[i] = (const float*)d_in[base + 2];
        gin_b2[i] = (const float*)d_in[base + 3];
        bn_g[i]   = (const float*)d_in[base + 4];
        bn_b[i]   = (const float*)d_in[base + 5];
    }
    const float* fc1_xd_w = (const float*)d_in[35];
    const float* fc1_xd_b = (const float*)d_in[36];
    const float* emb_xt   = (const float*)d_in[37];
    const float* conv1_w  = (const float*)d_in[38];
    const float* conv1_b  = (const float*)d_in[39];
    const float* se_w1    = (const float*)d_in[40];
    const float* se_b1    = (const float*)d_in[41];
    const float* se_w2    = (const float*)d_in[42];
    const float* se_b2    = (const float*)d_in[43];
    const float* fc_xt1_w = (const float*)d_in[44];
    const float* fc_xt1_b = (const float*)d_in[45];
    const float* emb_sm   = (const float*)d_in[46];
    const float* conv2_w  = (const float*)d_in[47];
    const float* conv2_b  = (const float*)d_in[48];
    const float* se1_w1   = (const float*)d_in[49];
    const float* se1_b1   = (const float*)d_in[50];
    const float* se1_w2   = (const float*)d_in[51];
    const float* se1_b2   = (const float*)d_in[52];
    const float* fc_xt2_w = (const float*)d_in[53];
    const float* fc_xt2_b = (const float*)d_in[54];
    const float* fc1_w    = (const float*)d_in[55];
    const float* fc1_b    = (const float*)d_in[56];
    const float* fc2_w    = (const float*)d_in[57];
    const float* fc2_b    = (const float*)d_in[58];
    const float* out_w    = (const float*)d_in[59];
    const float* out_b    = (const float*)d_in[60];

    // --- workspace carve ---
    char* base = (char*)d_ws;
    size_t cur = 0;
    auto alloc = [&](size_t bytes) {
        void* p = base + cur;
        cur += (bytes + 255) & ~(size_t)255;
        return p;
    };
    float*  agg    = (float*)alloc((size_t)N_NODES * FXD * 4);   // reused as N x 32 later
    float*  hA     = (float*)alloc((size_t)N_NODES * DIM * 4);
    float*  hB     = (float*)alloc((size_t)N_NODES * DIM * 4);
    float*  stats  = (float*)alloc(2 * DIM * 4);
    float*  g      = (float*)alloc((size_t)NB * DIM * 4);
    __bf16* gbf    = (__bf16*)alloc((size_t)NB * DIM * 2);
    float*  convbuf= (float*)alloc((size_t)NB * NF * CONVL * 4);
    float*  pooled = (float*)alloc((size_t)NB * NF * 4);
    __bf16* xtbf   = (__bf16*)alloc((size_t)NB * NF * CONVL * 2);
    __bf16* xdbf   = (__bf16*)alloc((size_t)NB * NF * CONVL * 2);
    float*  xc     = (float*)alloc((size_t)NB * 384 * 4);
    __bf16* xcbf   = (__bf16*)alloc((size_t)NB * 384 * 2);
    float*  buf1   = (float*)alloc((size_t)NB * 1024 * 4);
    __bf16* buf1bf = (__bf16*)alloc((size_t)NB * 1024 * 2);
    float*  buf2   = (float*)alloc((size_t)NB * 256 * 4);
    __bf16* wconvP = (__bf16*)alloc((size_t)NF * PLEN * 8 * 2);
    __bf16* wconvS = (__bf16*)alloc((size_t)NF * SLEN * 8 * 2);
    __bf16* wfcxt1 = (__bf16*)alloc((size_t)OUTD * NF * CONVL * 2);
    __bf16* wfcxt2 = (__bf16*)alloc((size_t)OUTD * NF * CONVL * 2);
    __bf16* wfc1   = (__bf16*)alloc((size_t)1024 * 384 * 2);
    __bf16* wfc2   = (__bf16*)alloc((size_t)256 * 1024 * 2);
    __bf16* wfc1xd = (__bf16*)alloc((size_t)OUTD * DIM * 2);

    auto cdiv = [](int a, int b) { return (a + b - 1) / b; };
    auto cvt = [&](const float* in, __bf16* out, int n) {
        cvt_bf16_kernel<<<cdiv(n, 256), 256, 0, stream>>>(in, out, n);
    };

    // --- weight conversions (recomputed every call; deterministic) ---
    cvt(conv1_w,  wconvP, NF * PLEN * 8);
    cvt(conv2_w,  wconvS, NF * SLEN * 8);
    cvt(fc_xt1_w, wfcxt1, OUTD * NF * CONVL);
    cvt(fc_xt2_w, wfcxt2, OUTD * NF * CONVL);
    cvt(fc1_w,    wfc1,   1024 * 384);
    cvt(fc2_w,    wfc2,   256 * 1024);
    cvt(fc1_xd_w, wfc1xd, OUTD * DIM);

    // --- GIN layer 0 (DIN=78) ---
    hipMemsetAsync(agg, 0, (size_t)N_NODES * FXD * 4, stream);
    edge_agg_kernel<FXD><<<cdiv(N_EDGES, 256), 256, 0, stream>>>(x, src, dst, agg, N_EDGES);
    hipMemsetAsync(stats, 0, 2 * DIM * 4, stream);
    gin_mlp_kernel<FXD><<<cdiv(N_NODES, 256), 256, 0, stream>>>(
        x, agg, gin_w1[0], gin_b1[0], gin_w2[0], gin_b2[0], hA, stats, N_NODES);
    bn_apply_kernel<<<cdiv(N_NODES * DIM, 256), 256, 0, stream>>>(hA, stats, bn_g[0], bn_b[0],
                                                                  N_NODES);
    // --- GIN layers 1..4 (DIN=32) ---
    float* hcur = hA;
    float* hnext = hB;
    for (int i = 1; i < 5; ++i) {
        hipMemsetAsync(agg, 0, (size_t)N_NODES * DIM * 4, stream);
        edge_agg_kernel<DIM><<<cdiv(N_EDGES, 256), 256, 0, stream>>>(hcur, src, dst, agg,
                                                                     N_EDGES);
        hipMemsetAsync(stats, 0, 2 * DIM * 4, stream);
        gin_mlp_kernel<DIM><<<cdiv(N_NODES, 256), 256, 0, stream>>>(
            hcur, agg, gin_w1[i], gin_b1[i], gin_w2[i], gin_b2[i], hnext, stats, N_NODES);
        bn_apply_kernel<<<cdiv(N_NODES * DIM, 256), 256, 0, stream>>>(hnext, stats, bn_g[i],
                                                                      bn_b[i], N_NODES);
        float* tmp = hcur; hcur = hnext; hnext = tmp;
    }

    // --- global add pool + fc1_xd (relu) into xc[:, 0:128] ---
    hipMemsetAsync(g, 0, (size_t)NB * DIM * 4, stream);
    pool_add_kernel<<<cdiv(N_NODES, 256), 256, 0, stream>>>(hcur, batch, g, N_NODES);
    cvt(g, gbf, NB * DIM);
    gemm_bf16_wmma<<<dim3(OUTD / 64, NB / 32), 256, 0, stream>>>(gbf, wfc1xd, fc1_xd_b, xc + 0,
                                                                 384, DIM, 1);

    // --- protein branch: conv + SE + fc_xt1 into xc[:, 128:256] ---
    hipMemsetAsync(pooled, 0, (size_t)NB * NF * 4, stream);
    conv_pool_wmma<<<NB, 256, 0, stream>>>(target, emb_xt, wconvP, conv1_b, convbuf, pooled,
                                           PLEN, 26);
    se_flatten_kernel<<<NB, 256, 0, stream>>>(convbuf, pooled, se_w1, se_b1, se_w2, se_b2, xtbf);
    gemm_bf16_wmma<<<dim3(OUTD / 64, NB / 32), 256, 0, stream>>>(xtbf, wfcxt1, fc_xt1_b,
                                                                 xc + 128, 384, NF * CONVL, 0);

    // --- smiles branch: conv + SE + fc_xt2 into xc[:, 256:384] ---
    hipMemsetAsync(pooled, 0, (size_t)NB * NF * 4, stream);
    conv_pool_wmma<<<NB, 256, 0, stream>>>(smiles, emb_sm, wconvS, conv2_b, convbuf, pooled,
                                           SLEN, 100);
    se_flatten_kernel<<<NB, 256, 0, stream>>>(convbuf, pooled, se1_w1, se1_b1, se1_w2, se1_b2,
                                              xdbf);
    gemm_bf16_wmma<<<dim3(OUTD / 64, NB / 32), 256, 0, stream>>>(xdbf, wfcxt2, fc_xt2_b,
                                                                 xc + 256, 384, NF * CONVL, 0);

    // --- head: fc1 (relu) -> fc2 (relu) -> out ---
    cvt(xc, xcbf, NB * 384);
    gemm_bf16_wmma<<<dim3(1024 / 64, NB / 32), 256, 0, stream>>>(xcbf, wfc1, fc1_b, buf1, 1024,
                                                                 384, 1);
    cvt(buf1, buf1bf, NB * 1024);
    gemm_bf16_wmma<<<dim3(256 / 64, NB / 32), 256, 0, stream>>>(buf1bf, wfc2, fc2_b, buf2, 256,
                                                                1024, 1);
    final_out_kernel<<<cdiv(NB, 256), 256, 0, stream>>>(buf2, out_w, out_b, (float*)d_out);
}
